// LoRALinear_58033598104232
// MI455X (gfx1250) — matrix operand order repaired and checked by reading, not compiled, not run on hardware
//
#include <hip/hip_runtime.h>
#include <stdint.h>

typedef __attribute__((ext_vector_type(16))) __bf16 v16bf;
typedef __attribute__((ext_vector_type(8)))  float  v8f;

#define D_IN    4096
#define D_OUT   4096
#define M_TOTAL (8 * 2048)
#define LORA_R  16
#define LORA_SCALE 2.0f

#define BM 128
#define BN 128
#define BK 32
#define LSTR 36            // padded LDS row stride (floats): 144B rows -> 16B aligned, bank-conflict-free
#define NK (D_IN / BK)     // 128 K-steps

// ---- CDNA5 async global->LDS copy (ASYNCcnt path), inline asm for toolchain portability ----
__device__ __forceinline__ void async_b128_to_lds(const float* gsrc, const float* lds_dst) {
    // low 32 bits of a flat pointer into LDS == LDS byte offset (aperture truncation)
    const unsigned lds_off = (unsigned)(uintptr_t)lds_dst;
    asm volatile("global_load_async_to_lds_b128 %0, %1, off"
                 :
                 : "v"(lds_off), "v"(gsrc)
                 : "memory");
}
__device__ __forceinline__ void wait_async0() {
    asm volatile("s_wait_asynccnt 0x0" ::: "memory");
}

__global__ __launch_bounds__(256)
void lora_gemm_bf16_wmma(const float* __restrict__ x,
                         const float* __restrict__ wb,
                         const float* __restrict__ bias,
                         const float* __restrict__ lA,
                         const float* __restrict__ lB,
                         float* __restrict__ out)
{
    __shared__ __align__(16) float  xs[2][BM][LSTR];
    __shared__ __align__(16) float  wsh[2][BN][LSTR];
    __shared__ __bf16 low_sh[BM][LORA_R];

    const int tid   = threadIdx.x;
    const int lane  = tid & 31;
    const int wid   = tid >> 5;
    const int wm    = wid & 3;       // wave row 0..3 -> 32 M rows each
    const int wn    = wid >> 2;      // wave col 0..1 -> 64 N cols each
    const int lrow  = lane & 15;
    const int lhalf = lane >> 4;

    const int blockM = blockIdx.y * BM;
    const int blockN = blockIdx.x * BN;

    // staging map: tile = 128 rows x 8 float4 columns; 4 async b128 per thread per matrix
    const int ldRow0 = tid >> 3;       // 0..31
    const int ldC4   = (tid & 7) * 4;  // float offset of the 16B column

    auto issue_tile = [&](int k0, int buf) {
#pragma unroll
        for (int i = 0; i < 4; ++i) {
            const int row = ldRow0 + 32 * i;
            async_b128_to_lds(x  + (size_t)(blockM + row) * D_IN + k0 + ldC4, &xs[buf][row][ldC4]);
            async_b128_to_lds(wb + (size_t)(blockN + row) * D_IN + k0 + ldC4, &wsh[buf][row][ldC4]);
        }
        if (k0 + BK < D_IN) {   // prefetch the tile after the one being copied
            __builtin_prefetch(x  + (size_t)(blockM + ldRow0) * D_IN + k0 + BK + ldC4, 0, 1);
            __builtin_prefetch(wb + (size_t)(blockN + ldRow0) * D_IN + k0 + BK + ldC4, 0, 1);
        }
    };

    // ---- accumulators ----
    const v8f vz = {0.f, 0.f, 0.f, 0.f, 0.f, 0.f, 0.f, 0.f};
    v8f acc[2][4];
    v8f lowacc[2];
#pragma unroll
    for (int mt = 0; mt < 2; ++mt) {
        lowacc[mt] = vz;
#pragma unroll
        for (int nt = 0; nt < 4; ++nt) acc[mt][nt] = vz;
    }

    // A fragment: 16x32 bf16, row = lrow; lanes 0-15 K0..15, lanes 16-31 K16..31
    auto make_a = [&](int buf, int rowBase) -> v16bf {
        v16bf f;
        const float* rp = &xs[buf][rowBase + lrow][0];
#pragma unroll
        for (int j = 0; j < 8; ++j) {
            const int k = lhalf * 16 + 2 * j;
            const float2 v = *reinterpret_cast<const float2*>(rp + k);
            f[2 * j]     = (__bf16)v.x;
            f[2 * j + 1] = (__bf16)v.y;
        }
        return f;
    };
    // B fragment: 32x16 bf16, col = lrow; lanes 0-15 K0..15, lanes 16-31 K16..31
    auto make_b = [&](int buf, int colBase) -> v16bf {
        v16bf f;
        const float* rp = &wsh[buf][colBase + lrow][0];
#pragma unroll
        for (int j = 0; j < 8; ++j) {
            const int k = lhalf * 16 + 2 * j;
            const float2 v = *reinterpret_cast<const float2*>(rp + k);
            f[2 * j]     = (__bf16)v.x;
            f[2 * j + 1] = (__bf16)v.y;
        }
        return f;
    };

    // prologue: async-copy tile 0 into buffer 0
    issue_tile(0, 0);
    wait_async0();
    __syncthreads();

    for (int ks = 0; ks < NK; ++ks) {
        const int  k0       = ks * BK;
        const int  cur      = ks & 1;
        const bool has_next = (ks + 1 < NK);

        // kick off the next tile's async copies; they run under the WMMA burst below
        if (has_next) issue_tile(k0 + BK, cur ^ 1);

        v16bf af[2];
#pragma unroll
        for (int mt = 0; mt < 2; ++mt) af[mt] = make_a(cur, wm * 32 + mt * 16);
        v16bf bfm[4];
#pragma unroll
        for (int nt = 0; nt < 4; ++nt) bfm[nt] = make_b(cur, wn * 64 + nt * 16);

#pragma unroll
        for (int mt = 0; mt < 2; ++mt)
#pragma unroll
            for (int nt = 0; nt < 4; ++nt)
                acc[mt][nt] = __builtin_amdgcn_wmma_f32_16x16x32_bf16(
                    false, af[mt], false, bfm[nt], (short)0, acc[mt][nt], false, false);

        // LoRA low-rank projection low = x * A^T fused into the same K loop.
        // wn is wave-uniform -> EXEC stays all-ones inside (WMMA requirement).
        if (wn == 0) {
            v16bf lb;
            const float* ap = lA + (size_t)lrow * D_IN + k0;   // B[k][r] = lora_A[r][k]
#pragma unroll
            for (int j = 0; j < 8; ++j) {
                const int k = lhalf * 16 + 2 * j;
                const float2 v = *reinterpret_cast<const float2*>(ap + k);
                lb[2 * j]     = (__bf16)v.x;
                lb[2 * j + 1] = (__bf16)v.y;
            }
#pragma unroll
            for (int mt = 0; mt < 2; ++mt)
                lowacc[mt] = __builtin_amdgcn_wmma_f32_16x16x32_bf16(
                    false, af[mt], false, lb, (short)0, lowacc[mt], false, false);
        }

        // make next buffer visible: own copies done (ASYNCcnt==0), then cross-wave barrier
        if (has_next) wait_async0();
        __syncthreads();
    }

    // ---- LoRA epilogue: share low via LDS, fold scale*low*B^T into acc ----
    if (wn == 0) {
#pragma unroll
        for (int mt = 0; mt < 2; ++mt) {
            const int rowBase = wm * 32 + mt * 16;
#pragma unroll
            for (int v = 0; v < 8; ++v)   // C layout: VGPR v -> M = 8*lhalf + v, N = lrow
                low_sh[rowBase + lhalf * 8 + v][lrow] = (__bf16)(LORA_SCALE * lowacc[mt][v]);
        }
    }
    __syncthreads();

    // low as A fragment, K padded 16->32 with zeros
    v16bf la[2];
#pragma unroll
    for (int mt = 0; mt < 2; ++mt) {
        const int rowBase = wm * 32 + mt * 16;
        v16bf f;
#pragma unroll
        for (int j = 0; j < 4; ++j) {
            const int k = lhalf * 8 + 2 * j;           // 0..15 = r index
            f[2 * j]     = low_sh[rowBase + lrow][k];
            f[2 * j + 1] = low_sh[rowBase + lrow][k + 1];
        }
#pragma unroll
        for (int j = 4; j < 8; ++j) { f[2 * j] = (__bf16)0.f; f[2 * j + 1] = (__bf16)0.f; }
        la[mt] = f;
    }

#pragma unroll
    for (int nt = 0; nt < 4; ++nt) {
        const int ng = blockN + wn * 64 + nt * 16 + lrow;
        v16bf fb;
#pragma unroll
        for (int j = 0; j < 8; ++j) {
            const int k = ((j < 4) ? 0 : 16) + lhalf * 8 + (j & 3) * 2;  // >=16 for j>=4 -> zero pad
            const float v0 = (k     < LORA_R) ? lB[(size_t)ng * LORA_R + k]     : 0.f;
            const float v1 = (k + 1 < LORA_R) ? lB[(size_t)ng * LORA_R + k + 1] : 0.f;
            fb[2 * j]     = (__bf16)v0;
            fb[2 * j + 1] = (__bf16)v1;
        }
#pragma unroll
        for (int mt = 0; mt < 2; ++mt)
            acc[mt][nt] = __builtin_amdgcn_wmma_f32_16x16x32_bf16(
                false, la[mt], false, fb, (short)0, acc[mt][nt], false, false);
    }

    // ---- bias + store ----
#pragma unroll
    for (int nt = 0; nt < 4; ++nt) {
        const int   ng = blockN + wn * 64 + nt * 16 + lrow;
        const float bv = bias[ng];
#pragma unroll
        for (int mt = 0; mt < 2; ++mt) {
            const int mBase = blockM + wm * 32 + mt * 16 + lhalf * 8;
#pragma unroll
            for (int v = 0; v < 8; ++v)
                out[(size_t)(mBase + v) * D_OUT + ng] = acc[mt][nt][v] + bv;
        }
    }
}

extern "C" void kernel_launch(void* const* d_in, const int* in_sizes, int n_in,
                              void* d_out, int out_size, void* d_ws, size_t ws_size,
                              hipStream_t stream) {
    (void)in_sizes; (void)n_in; (void)d_ws; (void)ws_size; (void)out_size;
    const float* x  = (const float*)d_in[0];
    const float* wb = (const float*)d_in[1];
    const float* b  = (const float*)d_in[2];
    const float* lA = (const float*)d_in[3];
    const float* lB = (const float*)d_in[4];
    float* out = (float*)d_out;

    dim3 grid(D_OUT / BN, M_TOTAL / BM);   // 32 x 128 workgroups
    dim3 block(256);                        // 8 wave32 waves
    hipLaunchKernelGGL(lora_gemm_bf16_wmma, grid, block, 0, stream,
                       x, wb, b, lA, lB, out);
}
